// Decoder_14139032338987
// MI455X (gfx1250) — matrix operand order repaired and checked by reading, not compiled, not run on hardware
//
#include <hip/hip_runtime.h>
#include <hip/hip_bf16.h>
#include <stdint.h>

// Problem constants (from reference)
#define H_DIM   1024
#define V_DIM   32000
#define NLAYER  4
#define WIN     10
#define S_LEN   512
#define BATCH   128
#define T_DIM   (S_LEN + 2*WIN)   // 532
#define L_DIM   (2*WIN + 1)       // 21
#define HALF_DIM 512
#define STD2_F  25.0f             // (W/2)^2

typedef __attribute__((ext_vector_type(16))) __bf16 v16bf;
typedef __attribute__((ext_vector_type(8)))  float  v8f;

union BfVec {
  v16bf v;
  uint32_t u[8];
  uint4 q[2];
};

// fp32 -> bf16 packed pair (round-half-up) via v_perm_b32: 3 VALU per pair
#if __has_builtin(__builtin_amdgcn_perm)
__device__ __forceinline__ uint32_t pk_bf16(float a, float b) {
  uint32_t ua = __float_as_uint(a) + 0x8000u;
  uint32_t ub = __float_as_uint(b) + 0x8000u;
  // result bytes: [0]=ua.b2 [1]=ua.b3 [2]=ub.b2 [3]=ub.b3  -> {bf(b), bf(a)}
  return __builtin_amdgcn_perm(ub, ua, 0x07060302u);
}
#else
__device__ __forceinline__ uint32_t pk_bf16(float a, float b) {
  uint32_t ua = (__float_as_uint(a) + 0x8000u) >> 16;
  uint32_t ub = (__float_as_uint(b) + 0x8000u) >> 16;
  return (ua & 0xFFFFu) | (ub << 16);
}
#endif

__device__ __forceinline__ unsigned short f2bf(float a) {
  return (unsigned short)((__float_as_uint(a) + 0x8000u) >> 16);
}

__device__ __forceinline__ float sigf(float x) { return 1.0f / (1.0f + expf(-x)); }

// ---------------------------------------------------------------------------
// WMMA GEMM: Y[M=128, N] = act( A1[M,K]*W1[N,K]^T (+ A2[M,K]*W2[N,K]^T) + bias )
//   A panels: bf16 (row-major, produced once by upstream kernels)
//   W panels: fp32 streamed from HBM, cvt to bf16 in registers (v_perm pack)
//   A k-blocks staged global->LDS with global_load_async_to_lds_b128 (ASYNCcnt),
//   overlapped with the W loads; consumed via ds_load_b128 in WMMA A layout,
//   software-pipelined one M-tile ahead of the WMMAs.
// Block = 128 threads = 4 waves; wave w owns N-tile (blockIdx.x*64 + w*16) and
// all 8 M-tiles. ACT: 0=none, 1=tanh. OUTBF: store bf16 instead of f32.
// ---------------------------------------------------------------------------
template<bool DUAL, int ACT, bool OUTBF>
__global__ __launch_bounds__(128) void wmma_gemm_kernel(
    const unsigned short* __restrict__ A1, const float* __restrict__ W1,
    const unsigned short* __restrict__ A2, const float* __restrict__ W2,
    const float* __restrict__ bias1, const float* __restrict__ bias2,
    void* __restrict__ Yv, int K, int N)
{
  // 2 streams x 128 rows x 32 bf16 (64 B) per row = 16 KB
  __shared__ __align__(16) uint32_t ldsA[2][128 * 16];

  const int t    = threadIdx.x;
  const int lane = t & 31;
  const int wave = t >> 5;
  const int n0   = blockIdx.x * 64 + wave * 16;
  const int nl   = lane & 15;
  const int kh   = (lane >> 4) ? 16 : 0;       // B-operand K-half for this lane

  const float* wrow1 = W1 + (size_t)(n0 + nl) * (size_t)K;
  const float* wrow2 = DUAL ? (W2 + (size_t)(n0 + nl) * (size_t)K) : nullptr;

  // Low 32 bits of a generic LDS pointer are the LDS byte offset (flat aperture).
  const unsigned lb0 = (unsigned)(uintptr_t)(void*)&ldsA[0][0];
  const unsigned lb1 = (unsigned)(uintptr_t)(void*)&ldsA[1][0];

  v8f acc[8] = {};

  for (int k = 0; k < K; k += 32) {
    __syncthreads();  // previous block fully consumed before refill

    // ---- async-stage A k-block(s): 128 rows x 64 B; 16 B per lane-op
    #pragma unroll
    for (int i = 0; i < 4; ++i) {
      int e   = i * 128 + t;
      int row = e >> 2, seg = e & 3;
      unsigned    ld = lb0 + (unsigned)(row * 64 + seg * 16);
      const char* ga = (const char*)A1 + ((size_t)row * K + k) * 2 + seg * 16;
      asm volatile("global_load_async_to_lds_b128 %0, %1, off"
                   :: "v"(ld), "v"(ga) : "memory");
    }
    if (DUAL) {
      #pragma unroll
      for (int i = 0; i < 4; ++i) {
        int e   = i * 128 + t;
        int row = e >> 2, seg = e & 3;
        unsigned    ld = lb1 + (unsigned)(row * 64 + seg * 16);
        const char* ga = (const char*)A2 + ((size_t)row * K + k) * 2 + seg * 16;
        asm volatile("global_load_async_to_lds_b128 %0, %1, off"
                     :: "v"(ld), "v"(ga) : "memory");
      }
    }

    // ---- B operands (overlap with the async LDS fill): 16 fp32 -> bf16
    // lane<16: N=nl, K=k+0..15 ; lane>=16: N=nl, K=k+16..31
    BfVec b1;
    {
      const float4* wp = (const float4*)(wrow1 + k + kh);
      __builtin_prefetch(wrow1 + k + 128, 0, 0);   // 4 k-blocks ahead -> GL2
      #pragma unroll
      for (int i = 0; i < 4; ++i) {
        float4 v = wp[i];
        b1.u[2*i+0] = pk_bf16(v.x, v.y);
        b1.u[2*i+1] = pk_bf16(v.z, v.w);
      }
    }
    BfVec b2;
    if (DUAL) {
      const float4* wp = (const float4*)(wrow2 + k + kh);
      __builtin_prefetch(wrow2 + k + 128, 0, 0);
      #pragma unroll
      for (int i = 0; i < 4; ++i) {
        float4 v = wp[i];
        b2.u[2*i+0] = pk_bf16(v.x, v.y);
        b2.u[2*i+1] = pk_bf16(v.z, v.w);
      }
    }

    asm volatile("s_wait_asynccnt 0x0" ::: "memory");
    __syncthreads();

    // A operand layout: lanes 0-15 hold K{0..15} (dwords 0..7)
    //                   lanes 16-31 hold K{16..31} (dwords 8..15)
    const int doff = (lane < 16) ? 0 : 8;
    const uint32_t* abase0 = &ldsA[0][nl * 16];
    const uint32_t* abase1 = &ldsA[1][nl * 16];

    BfVec a0, a1;
    a0.q[0] = *(const uint4*)(abase0 + doff);
    a0.q[1] = *(const uint4*)(abase0 + doff + 4);
    if (DUAL) {
      a1.q[0] = *(const uint4*)(abase1 + doff);
      a1.q[1] = *(const uint4*)(abase1 + doff + 4);
    }

    #pragma unroll
    for (int mt = 0; mt < 8; ++mt) {
      BfVec an0, an1;
      if (mt < 7) {  // prefetch next M-tile while current WMMAs issue
        const uint32_t* ar0 = abase0 + (mt + 1) * 256;
        an0.q[0] = *(const uint4*)(ar0 + doff);
        an0.q[1] = *(const uint4*)(ar0 + doff + 4);
        if (DUAL) {
          const uint32_t* ar1 = abase1 + (mt + 1) * 256;
          an1.q[0] = *(const uint4*)(ar1 + doff);
          an1.q[1] = *(const uint4*)(ar1 + doff + 4);
        }
      }
      acc[mt] = __builtin_amdgcn_wmma_f32_16x16x32_bf16(
          false, a0.v, false, b1.v, (short)0, acc[mt], false, false);
      if (DUAL) {
        acc[mt] = __builtin_amdgcn_wmma_f32_16x16x32_bf16(
            false, a1.v, false, b2.v, (short)0, acc[mt], false, false);
      }
      a0 = an0;
      if (DUAL) a1 = an1;
    }
  }

  // ---- epilogue: C/D layout lanes0-15: M=r, lanes16-31: M=r+8 (N = nl)
  float bv = bias1[n0 + nl];
  if (DUAL) bv += bias2[n0 + nl];
  const int madd = (lane < 16) ? 0 : 8;
  #pragma unroll
  for (int mt = 0; mt < 8; ++mt) {
    #pragma unroll
    for (int r = 0; r < 8; ++r) {
      int m = mt * 16 + r + madd;
      float v = acc[mt][r] + bv;
      if (ACT == 1) v = tanhf(v);
      if (OUTBF)
        ((unsigned short*)Yv)[(size_t)m * N + n0 + nl] = f2bf(v);
      else
        ((float*)Yv)[(size_t)m * N + n0 + nl] = v;
    }
  }
}

// ---------------------------------------------------------------------------
// Embedding gather -> bf16 panel: x[b,:] = bf16(emb[word[b],:])
// ---------------------------------------------------------------------------
__global__ __launch_bounds__(256) void embed_kernel(
    const float* __restrict__ emb, const int* __restrict__ word,
    unsigned short* __restrict__ xb)
{
  int b = blockIdx.x, t = threadIdx.x;
  int w = word[b];
  const float4* src = (const float4*)(emb + (size_t)w * H_DIM);
  float4 v = src[t];
  uint2 o;
  o.x = pk_bf16(v.x, v.y);
  o.y = pk_bf16(v.z, v.w);
  ((uint2*)(xb + (size_t)b * H_DIM))[t] = o;
}

// ---------------------------------------------------------------------------
// fp32 -> bf16 panel conversion (for h0)
// ---------------------------------------------------------------------------
__global__ __launch_bounds__(256) void cvt_bf16_kernel(
    const float* __restrict__ src, unsigned short* __restrict__ dst, int n4)
{
  int i = blockIdx.x * blockDim.x + threadIdx.x;
  if (i >= n4) return;
  float4 v = ((const float4*)src)[i];
  uint2 o;
  o.x = pk_bf16(v.x, v.y);
  o.y = pk_bf16(v.z, v.w);
  ((uint2*)dst)[i] = o;
}

// ---------------------------------------------------------------------------
// LSTM cell: gates[b,4H] (i,f,g,o) + c0 -> h, written fp32 + bf16
// ---------------------------------------------------------------------------
__global__ __launch_bounds__(256) void lstm_cell_kernel(
    const float* __restrict__ gates, const float* __restrict__ c0,
    float* __restrict__ houtf, unsigned short* __restrict__ houtb)
{
  int idx = blockIdx.x * blockDim.x + threadIdx.x;
  if (idx >= BATCH * H_DIM) return;
  int b = idx >> 10, h = idx & (H_DIM - 1);
  const float* g = gates + (size_t)b * 4 * H_DIM;
  float ig = g[h];
  float fg = g[H_DIM + h];
  float gg = g[2 * H_DIM + h];
  float og = g[3 * H_DIM + h];
  float c  = sigf(fg) * c0[idx] + sigf(ig) * tanhf(gg);
  float hv = sigf(og) * tanhf(c);
  houtf[idx] = hv;
  houtb[idx] = f2bf(hv);
}

// ---------------------------------------------------------------------------
// p[b] = W + S * sigmoid( pt[b,:] . aW2 + ab2 )
// ---------------------------------------------------------------------------
__global__ __launch_bounds__(128) void attn_p_kernel(
    const float* __restrict__ pt, const float* __restrict__ aW2,
    const float* __restrict__ ab2, float* __restrict__ p)
{
  __shared__ float red[128];
  int b = blockIdx.x, t = threadIdx.x;
  float s = 0.0f;
  for (int j = t; j < HALF_DIM; j += 128)
    s += pt[(size_t)b * HALF_DIM + j] * aW2[j];
  red[t] = s; __syncthreads();
  for (int off = 64; off > 0; off >>= 1) {
    if (t < off) red[t] += red[t + off];
    __syncthreads();
  }
  if (t == 0) {
    float v = red[0] + ab2[0];
    p[b] = (float)WIN + (float)S_LEN * sigf(v);
  }
}

// ---------------------------------------------------------------------------
// Windowed attention: scores, softmax*gaussian, context; writes bf16
// cat = [ctx, out]. One workgroup (256 threads) per batch row.
// ---------------------------------------------------------------------------
__global__ __launch_bounds__(256) void attn_ctx_kernel(
    const float* __restrict__ enc, const float* __restrict__ out,
    const float* __restrict__ p, unsigned short* __restrict__ catb)
{
  __shared__ float red[256];
  __shared__ float aw[L_DIM];
  int b = blockIdx.x, t = threadIdx.x;
  float pb = p[b];
  int start = (int)rintf(pb - (float)WIN);

  const float* ob = out + (size_t)b * H_DIM;

  for (int l = 0; l < L_DIM; ++l) {
    int pos  = start + l;
    int cpos = pos < 0 ? 0 : (pos > T_DIM - 1 ? T_DIM - 1 : pos);
    const float* sel = enc + (size_t)cpos * BATCH * H_DIM + (size_t)b * H_DIM;
    float s = 0.0f;
    for (int h = t; h < H_DIM; h += 256) s += ob[h] * sel[h];
    red[t] = s; __syncthreads();
    for (int off = 128; off > 0; off >>= 1) {
      if (t < off) red[t] += red[t + off];
      __syncthreads();
    }
    if (t == 0) aw[l] = red[0];
    __syncthreads();
  }

  if (t == 0) {
    float sc[L_DIM];
    float mx = -1e30f;
    for (int l = 0; l < L_DIM; ++l) {
      int pos = start + l;
      bool valid = (pos >= WIN) && (pos < S_LEN + WIN);
      float s = valid ? aw[l] : 1e-12f;
      sc[l] = s;
      mx = fmaxf(mx, s);
    }
    float sum = 0.0f;
    for (int l = 0; l < L_DIM; ++l) { sc[l] = expf(sc[l] - mx); sum += sc[l]; }
    float inv = 1.0f / sum;
    for (int l = 0; l < L_DIM; ++l) {
      float d = (float)(start + l) - pb;
      aw[l] = sc[l] * inv * expf(-d * d / (2.0f * STD2_F));
    }
  }
  __syncthreads();

  unsigned short* cb = catb + (size_t)b * (2 * H_DIM);
  for (int h = t; h < H_DIM; h += 256) {
    float c = 0.0f;
    for (int l = 0; l < L_DIM; ++l) {
      int pos  = start + l;
      int cpos = pos < 0 ? 0 : (pos > T_DIM - 1 ? T_DIM - 1 : pos);
      c += aw[l] * enc[(size_t)cpos * BATCH * H_DIM + (size_t)b * H_DIM + h];
    }
    cb[h] = f2bf(c);
    cb[H_DIM + h] = f2bf(ob[h]);
  }
}

// ---------------------------------------------------------------------------
extern "C" void kernel_launch(void* const* d_in, const int* in_sizes, int n_in,
                              void* d_out, int out_size, void* d_ws, size_t ws_size,
                              hipStream_t stream) {
  (void)in_sizes; (void)n_in; (void)out_size; (void)ws_size;

  const float* enc  = (const float*)d_in[0];   // (T, B, H)
  const float* h0   = (const float*)d_in[1];   // (NL, B, H)
  const float* c0   = (const float*)d_in[2];   // (NL, B, H)
  const float* emb  = (const float*)d_in[3];   // (V, H)
  const float* Wih  = (const float*)d_in[4];   // (NL, 4H, H)
  const float* Whh  = (const float*)d_in[5];   // (NL, 4H, H)
  const float* bih  = (const float*)d_in[6];   // (NL, 4H)
  const float* bhh  = (const float*)d_in[7];   // (NL, 4H)
  const float* aW1  = (const float*)d_in[8];   // (HALF, H)
  const float* ab1  = (const float*)d_in[9];   // (HALF,)
  const float* aW2  = (const float*)d_in[10];  // (1, HALF)
  const float* ab2  = (const float*)d_in[11];  // (1,)
  const float* dW1  = (const float*)d_in[12];  // (H, 2H)
  const float* db1  = (const float*)d_in[13];  // (H,)
  const float* dW2  = (const float*)d_in[14];  // (V, H)
  const float* db2  = (const float*)d_in[15];  // (V,)
  const int*   word = (const int*)d_in[16];    // (1, B)

  // workspace carve (fp32 first, then bf16 panels)
  char* w = (char*)d_ws;
  float* gates = (float*)w;              w += (size_t)BATCH * 4 * H_DIM * 4;
  float* pt    = (float*)w;              w += (size_t)BATCH * HALF_DIM * 4;
  float* hf    = (float*)w;              w += (size_t)BATCH * H_DIM * 4;
  float* pv    = (float*)w;              w += 256 * 4;
  unsigned short* x0b  = (unsigned short*)w; w += (size_t)BATCH * H_DIM * 2;
  unsigned short* h0b  = (unsigned short*)w; w += (size_t)NLAYER * BATCH * H_DIM * 2;
  unsigned short* hb   = (unsigned short*)w; w += (size_t)BATCH * H_DIM * 2;
  unsigned short* catb = (unsigned short*)w; w += (size_t)BATCH * 2 * H_DIM * 2;
  unsigned short* o2b  = (unsigned short*)w; w += (size_t)BATCH * H_DIM * 2;

  // 1) embedding gather (bf16) + h0 -> bf16 panels
  embed_kernel<<<BATCH, 256, 0, stream>>>(emb, word, x0b);
  {
    int n4 = NLAYER * BATCH * H_DIM / 4;
    cvt_bf16_kernel<<<(n4 + 255) / 256, 256, 0, stream>>>(h0, h0b, n4);
  }

  // 2) LSTM stack (sequential layer dependency)
  const unsigned short* ain = x0b;
  for (int l = 0; l < NLAYER; ++l) {
    wmma_gemm_kernel<true, 0, false><<<(4 * H_DIM) / 64, 128, 0, stream>>>(
        ain, Wih + (size_t)l * 4 * H_DIM * H_DIM,
        h0b + (size_t)l * BATCH * H_DIM, Whh + (size_t)l * 4 * H_DIM * H_DIM,
        bih + (size_t)l * 4 * H_DIM, bhh + (size_t)l * 4 * H_DIM,
        gates, H_DIM, 4 * H_DIM);
    lstm_cell_kernel<<<(BATCH * H_DIM + 255) / 256, 256, 0, stream>>>(
        gates, c0 + (size_t)l * BATCH * H_DIM, hf, hb);
    ain = hb;
  }
  // hf/hb now hold `out` (fp32 / bf16)

  // 3) pt = tanh(out @ aW1^T + ab1)
  wmma_gemm_kernel<false, 1, false><<<HALF_DIM / 64, 128, 0, stream>>>(
      hb, aW1, nullptr, nullptr, ab1, nullptr, pt, H_DIM, HALF_DIM);

  // 4) p = W + S*sigmoid(pt @ aW2^T + ab2)
  attn_p_kernel<<<BATCH, 128, 0, stream>>>(pt, aW2, ab2, pv);

  // 5) windowed attention -> cat = [ctx, out] (bf16)
  attn_ctx_kernel<<<BATCH, 256, 0, stream>>>(enc, hf, pv, catb);

  // 6) o2 = tanh(cat @ dW1^T + db1) -> bf16 panel
  wmma_gemm_kernel<false, 1, true><<<H_DIM / 64, 128, 0, stream>>>(
      catb, dW1, nullptr, nullptr, db1, nullptr, o2b, 2 * H_DIM, H_DIM);

  // 7) y = o2 @ dW2^T + db2 (fp32, straight into d_out)
  wmma_gemm_kernel<false, 0, false><<<V_DIM / 64, 128, 0, stream>>>(
      o2b, dW2, nullptr, nullptr, db2, nullptr, (float*)d_out, H_DIM, V_DIM);
}
